// Model_68607807586970
// MI455X (gfx1250) — matrix operand order repaired and checked by reading, not compiled, not run on hardware
//
#include <hip/hip_runtime.h>
#include <hip/hip_bf16.h>
#include <stdint.h>

// ---------------- model dims ----------------
#define H_   768
#define L_   2
#define NH_  12
#define DH_  64
#define FF_  3072
#define B_   64
#define S_   512
#define M_   (B_*S_)   // 32768 token rows

typedef __attribute__((ext_vector_type(16))) _Float16 v16h;
typedef __attribute__((ext_vector_type(8)))  float    v8f;
typedef __attribute__((ext_vector_type(4)))  unsigned u32x4;
typedef __attribute__((ext_vector_type(8)))  unsigned u32x8;

union V16U { uint4 u4[2]; v16h h; };

// A-matrix 16x32 f16 fragment: lane row = L&15, half hh = L>>4,
// element e -> K = 16*hh + e  => one contiguous 16-half run.
// p must point at the k = 16*hh element.
__device__ __forceinline__ v16h load_a_frag(const _Float16* p) {
  V16U t; t.u4[0] = ((const uint4*)p)[0]; t.u4[1] = ((const uint4*)p)[1]; return t.h;
}
// B-matrix 32x16 f16 fragment: lane col = L&15, element e -> K = 16*hh + e
// (one contiguous 16-half run).  p must point at the k = 16*hh element.
__device__ __forceinline__ v16h load_b_frag(const _Float16* p) {
  V16U t; t.u4[0] = ((const uint4*)p)[0]; t.u4[1] = ((const uint4*)p)[1]; return t.h;
}

__device__ __forceinline__ v8f wmma_f16(v16h a, v16h b, v8f c) {
  return __builtin_amdgcn_wmma_f32_16x16x32_f16(false, a, false, b, (short)0, c,
                                                false, false);
}

// low 32 bits of a flat LDS pointer == LDS byte offset (ISA 10.2 aperture map)
__device__ __forceinline__ unsigned lds_off(const void* p) {
  return (unsigned)(size_t)p;
}

// ---- Tensor Data Mover: DMA a [tile_rows x 32]-element f16 2D tile (row
// stride = K elements) from global memory into LDS, packed [tile_rows][32].
// Two-SGPR-group D# (2D tensor); tracked by TENSORcnt.
__device__ __forceinline__ void tdm_load_tile(const _Float16* gptr,
                                              unsigned lds_byte_off,
                                              int K_elems, int rows_total,
                                              unsigned tile_rows) {
  unsigned long long ga = (unsigned long long)(size_t)gptr;
  u32x4 g0; u32x8 g1;
  g0[0] = 1u;                                    // count=1, user mode
  g0[1] = lds_byte_off;                          // lds_addr
  g0[2] = (unsigned)ga;                          // global_addr[31:0]
  g0[3] = (unsigned)(ga >> 32) | 0x80000000u;    // global_addr[56:32] | type=2
  unsigned td0 = (unsigned)K_elems;              // tensor_dim0 (elements)
  unsigned td1 = (unsigned)rows_total;           // tensor_dim1
  g1[0] = 0x00010000u;                           // workgroup_mask=0, data_size=1 (2B)
  g1[1] = (td0 & 0xFFFFu) << 16;                 // tensor_dim0[15:0]
  g1[2] = (td0 >> 16) | ((td1 & 0xFFFFu) << 16); // tensor_dim0[31:16] | tensor_dim1[15:0]
  g1[3] = (td1 >> 16) | (32u << 16);             // tensor_dim1[31:16] | tile_dim0=32
  g1[4] = tile_rows;                             // tile_dim1, tile_dim2=0
  g1[5] = (unsigned)K_elems;                     // tensor_dim0_stride[31:0]
  g1[6] = 0u;                                    // stride0[47:32] | stride1[15:0]
  g1[7] = 0u;                                    // stride1[47:16]
  asm volatile("tensor_load_to_lds %0, %1" :: "s"(g0), "s"(g1) : "memory");
}

__device__ __forceinline__ float block_reduce_sum256(float v, float* red) {
  int tid = threadIdx.x;
  red[tid] = v; __syncthreads();
  for (int s = 128; s > 0; s >>= 1) { if (tid < s) red[tid] += red[tid + s]; __syncthreads(); }
  float r = red[0]; __syncthreads();
  return r;
}

__device__ __forceinline__ float gelu_exact(float x) {
  return 0.5f * x * (1.0f + erff(x * 0.70710678118654752f));
}

// ---------------- weight f32 -> f16 with transpose: dst[N][K] = src[K][N] ----
__global__ __launch_bounds__(256)
void convert_transpose_kernel(const float* __restrict__ src, _Float16* __restrict__ dst,
                              int K, int N) {
  __shared__ float tile[32][33];
  int k0 = blockIdx.y * 32, n0 = blockIdx.x * 32;
  int tx = threadIdx.x & 31, ty = threadIdx.x >> 5;   // 32 x 8
#pragma unroll
  for (int i = 0; i < 32; i += 8)
    tile[ty + i][tx] = src[(size_t)(k0 + ty + i) * N + n0 + tx];
  __syncthreads();
#pragma unroll
  for (int i = 0; i < 32; i += 8)
    dst[(size_t)(n0 + ty + i) * K + k0 + tx] = (_Float16)tile[tx][ty + i];
}

// ---------------- embedding + node average + LN ----------------
__global__ __launch_bounds__(256)
void embed_ln_kernel(const int* __restrict__ source_ids,
                     const int* __restrict__ position_idx,
                     const unsigned char* __restrict__ attn_mask,
                     const float* __restrict__ word_emb,
                     const float* __restrict__ pos_emb,
                     const float* __restrict__ ln_g,
                     const float* __restrict__ ln_b,
                     float* __restrict__ h32, _Float16* __restrict__ h16) {
  __shared__ float red[256];
  int row = blockIdx.x;              // b*S + s
  int b = row >> 9, s = row & (S_ - 1);
  int tid = threadIdx.x;
  int pos = position_idx[row];
  int id  = source_ids[row];
  float v[3];
#pragma unroll
  for (int j = 0; j < 3; ++j) v[j] = word_emb[(size_t)id * H_ + tid + j * 256];
  if (pos == 0) {  // node row: replace by masked average of token embeddings
    float acc[3] = {0.f, 0.f, 0.f}; float cnt = 0.f;
    const int* prow = position_idx + (size_t)b * S_;
    const int* irow = source_ids   + (size_t)b * S_;
    const unsigned char* mrow = attn_mask + ((size_t)b * S_ + s) * S_;
    for (int t = 0; t < S_; ++t) {
      if (prow[t] >= 2 && mrow[t]) {
        int idt = irow[t]; cnt += 1.f;
#pragma unroll
        for (int j = 0; j < 3; ++j) acc[j] += word_emb[(size_t)idt * H_ + tid + j * 256];
      }
    }
    float inv = 1.f / (cnt + 1e-10f);
#pragma unroll
    for (int j = 0; j < 3; ++j) v[j] = acc[j] * inv;
  }
#pragma unroll
  for (int j = 0; j < 3; ++j) v[j] += pos_emb[(size_t)pos * H_ + tid + j * 256];
  float mean = block_reduce_sum256(v[0] + v[1] + v[2], red) * (1.f / H_);
  float d0 = v[0] - mean, d1 = v[1] - mean, d2 = v[2] - mean;
  float var = block_reduce_sum256(d0 * d0 + d1 * d1 + d2 * d2, red) * (1.f / H_);
  float rinv = rsqrtf(var + 1e-5f);
#pragma unroll
  for (int j = 0; j < 3; ++j) {
    int c = tid + j * 256;
    float o = (v[j] - mean) * rinv * ln_g[c] + ln_b[c];
    h32[(size_t)row * H_ + c] = o;
    h16[(size_t)row * H_ + c] = (_Float16)o;
  }
}

// ---------------- WMMA GEMM, 64x128 block tile, TDM double-buffered --------
// out = A(f16,[M][K]) @ Wt(f16,[N][K])^T + bias
// 8 waves in a 4x2 grid; each wave computes 16 rows x 64 cols (4 accumulators)
// so one A-fragment feeds 4 WMMAs per K-step.
// MODE 0: store f16;  MODE 1: store f32;  MODE 2: store f16 of exact GELU.
template <int MODE>
__global__ __launch_bounds__(256)
void gemm_kernel(const _Float16* __restrict__ A, const _Float16* __restrict__ Wt,
                 const float* __restrict__ bias, float* __restrict__ out32,
                 _Float16* __restrict__ out16, int M, int N, int K) {
  __shared__ __align__(16) _Float16 As[2][64 * 32];    // [m][k]
  __shared__ __align__(16) _Float16 Bt[2][128 * 32];   // [n][k]
  int tid = threadIdx.x;
  int lane = tid & 31, wave = tid >> 5;
  int wm = wave >> 1, wn = wave & 1;                   // 4x2 wave grid
  int row0 = blockIdx.y * 64;
  int n0   = blockIdx.x * 128;
  int laneRow = lane & 15, hh = lane >> 4;
  int nsteps = K >> 5;

  v8f acc[4] = {};

  const _Float16* aptr = A  + (size_t)row0 * K;
  const _Float16* bptr = Wt + (size_t)n0 * K;

  if (wave == 0) {            // one wave drives the Tensor Data Mover
    tdm_load_tile(aptr, lds_off(&As[0][0]), K, M, 64u);
    tdm_load_tile(bptr, lds_off(&Bt[0][0]), K, N, 128u);
  }
  for (int i = 0; i < nsteps; ++i) {
    int p = i & 1;
    if (wave == 0) {
      if (i + 1 < nsteps) {   // prefetch next K-tile into the other buffer
        tdm_load_tile(aptr + (size_t)(i + 1) * 32, lds_off(&As[p ^ 1][0]), K, M, 64u);
        tdm_load_tile(bptr + (size_t)(i + 1) * 32, lds_off(&Bt[p ^ 1][0]), K, N, 128u);
        __builtin_amdgcn_s_wait_tensorcnt(2);   // pair(i) done (in-order)
      } else {
        __builtin_amdgcn_s_wait_tensorcnt(0);
      }
    }
    __syncthreads();
    v16h a = load_a_frag(&As[p][(wm * 16 + laneRow) * 32 + 16 * hh]);
#pragma unroll
    for (int j = 0; j < 4; ++j) {
      v16h b = load_b_frag(&Bt[p][(wn * 64 + j * 16 + laneRow) * 32 + 16 * hh]);
      acc[j] = wmma_f16(a, b, acc[j]);
    }
    __syncthreads();
  }
  int rowB = row0 + wm * 16 + hh * 8;
#pragma unroll
  for (int j = 0; j < 4; ++j) {
    int col = n0 + wn * 64 + j * 16 + laneRow;
    float bi = bias[col];
#pragma unroll
    for (int r = 0; r < 8; ++r) {
      size_t row = (size_t)(rowB + r);
      float v = acc[j][r] + bi;
      if (MODE == 0)      out16[row * N + col] = (_Float16)v;
      else if (MODE == 1) out32[row * N + col] = v;
      else                out16[row * N + col] = (_Float16)gelu_exact(v);
    }
  }
}

// ---------------- V transpose: vt[b,head,d,s] = qkv[b,s, 2H + head*64 + d] ----
__global__ __launch_bounds__(256)
void vtranspose_kernel(const _Float16* __restrict__ qkv, _Float16* __restrict__ vt) {
  int bh = blockIdx.x;                 // b*NH + head
  int b = bh / NH_, head = bh % NH_;
  const _Float16* vsrc = qkv + (size_t)b * S_ * 2304 + 2 * H_ + head * 64;
  _Float16* vdst = vt + (size_t)bh * 64 * 512;
  int t = threadIdx.x;
  int d  = t >> 2;                     // 0..63
  int s0 = (t & 3) * 128;              // write-contiguous chunks
  for (int i = 0; i < 128; ++i) {
    int s = s0 + i;
    vdst[(size_t)d * 512 + s] = vsrc[(size_t)s * 2304 + d];
  }
}

// ---------------- attention: one wave per (b, head, 16-query tile) ----------
__global__ __launch_bounds__(32)
void attention_kernel(const _Float16* __restrict__ qkv,       // [M, 2304] f16
                      const _Float16* __restrict__ vt,        // [B,NH,64,512] f16
                      const unsigned char* __restrict__ mask, // [B,S,S] bool
                      _Float16* __restrict__ ctx) {           // [M, 768] f16
  __shared__ __align__(16) float    sc[16 * 512];
  __shared__ __align__(16) _Float16 at[16 * 512];
  int lane = threadIdx.x;
  int qt   = blockIdx.x & 31;
  int head = (blockIdx.x >> 5) % NH_;
  int b    = blockIdx.x / (32 * NH_);
  int laneRow = lane & 15, hh = lane >> 4;
  size_t base = (size_t)b * S_ * 2304;

  // Q fragments (A-matrix 16x64 over DH, two K=32 steps)
  const _Float16* qb = qkv + base + (size_t)(qt * 16 + laneRow) * 2304 + head * 64;
  v16h qa0 = load_a_frag(qb + 16 * hh);
  v16h qa1 = load_a_frag(qb + 32 + 16 * hh);

  const unsigned char* mrow = mask + (size_t)b * S_ * S_;
  // scores = Q K^T * scale + mask bias
  for (int kt = 0; kt < 32; ++kt) {
    const _Float16* kb = qkv + base + (size_t)(kt * 16 + laneRow) * 2304 + H_ + head * 64;
    v16h kb0 = load_b_frag(kb + 16 * hh);        // B lane owns key column, contiguous d
    v16h kb1 = load_b_frag(kb + 32 + 16 * hh);
    v8f c = {};
    c = wmma_f16(qa0, kb0, c);
    c = wmma_f16(qa1, kb1, c);
#pragma unroll
    for (int r = 0; r < 8; ++r) {
      int m = r + hh * 8;
      int col = kt * 16 + laneRow;
      float bia = mrow[(size_t)(qt * 16 + m) * S_ + col] ? 0.f : -10000.f;
      sc[m * 512 + col] = c[r] * 0.125f + bia;   // scale = 1/sqrt(64)
    }
  }
  asm volatile("s_wait_dscnt 0" ::: "memory");
  // row softmax (wave-cooperative, shuffle reductions)
  for (int m = 0; m < 16; ++m) {
    float vals[16]; float mx = -3.4e38f;
#pragma unroll
    for (int i = 0; i < 16; ++i) { vals[i] = sc[m * 512 + lane + i * 32]; mx = fmaxf(mx, vals[i]); }
#pragma unroll
    for (int off = 16; off > 0; off >>= 1) mx = fmaxf(mx, __shfl_xor(mx, off, 32));
    float sum = 0.f;
#pragma unroll
    for (int i = 0; i < 16; ++i) { vals[i] = __expf(vals[i] - mx); sum += vals[i]; }
#pragma unroll
    for (int off = 16; off > 0; off >>= 1) sum += __shfl_xor(sum, off, 32);
    float inv = 1.f / sum;
#pragma unroll
    for (int i = 0; i < 16; ++i) at[m * 512 + lane + i * 32] = (_Float16)(vals[i] * inv);
  }
  asm volatile("s_wait_dscnt 0" ::: "memory");
  // ctx = attn @ V   (K = 512, 16 WMMA steps, 4 output d-tiles)
  for (int nt = 0; nt < 4; ++nt) {
    v8f c = {};
    const _Float16* vb = vt + ((size_t)(b * NH_ + head) * 64 + nt * 16 + laneRow) * 512;
    for (int kt = 0; kt < 16; ++kt) {
      v16h a  = load_a_frag(&at[laneRow * 512 + kt * 32 + 16 * hh]);
      v16h bf = load_b_frag(vb + kt * 32 + 16 * hh);
      c = wmma_f16(a, bf, c);
    }
#pragma unroll
    for (int r = 0; r < 8; ++r) {
      int m = r + hh * 8;
      ctx[((size_t)b * S_ + qt * 16 + m) * H_ + head * 64 + nt * 16 + laneRow] = (_Float16)c[r];
    }
  }
}

// ---------------- residual + LayerNorm ----------------
__global__ __launch_bounds__(256)
void ln_residual_kernel(float* __restrict__ h32, _Float16* __restrict__ h16,
                        const float* __restrict__ proj,
                        const float* __restrict__ g, const float* __restrict__ bb) {
  __shared__ float red[256];
  int row = blockIdx.x, tid = threadIdx.x;
  float v[3];
#pragma unroll
  for (int j = 0; j < 3; ++j) {
    int c = tid + j * 256;
    v[j] = h32[(size_t)row * H_ + c] + proj[(size_t)row * H_ + c];
  }
  float mean = block_reduce_sum256(v[0] + v[1] + v[2], red) * (1.f / H_);
  float d0 = v[0] - mean, d1 = v[1] - mean, d2 = v[2] - mean;
  float var = block_reduce_sum256(d0 * d0 + d1 * d1 + d2 * d2, red) * (1.f / H_);
  float rinv = rsqrtf(var + 1e-5f);
#pragma unroll
  for (int j = 0; j < 3; ++j) {
    int c = tid + j * 256;
    float o = (v[j] - mean) * rinv * g[c] + bb[c];
    h32[(size_t)row * H_ + c] = o;
    h16[(size_t)row * H_ + c] = (_Float16)o;
  }
}

// ---------------- pooled head: tanh(h[:,0] @ dW + db) @ oW + ob ----------------
__global__ __launch_bounds__(256)
void head_kernel(const float* __restrict__ h32, const float* __restrict__ dW,
                 const float* __restrict__ db, const float* __restrict__ oW,
                 const float* __restrict__ ob, float* __restrict__ out) {
  __shared__ float red[256];
  int b = blockIdx.x, tid = threadIdx.x;
  const float* x = h32 + (size_t)b * S_ * H_;   // row s = 0
  float part = 0.f;
#pragma unroll
  for (int j = 0; j < 3; ++j) {
    int col = tid + j * 256;
    float a = 0.f;
    for (int k = 0; k < H_; ++k) a += x[k] * dW[(size_t)k * H_ + col];
    part += tanhf(a + db[col]) * oW[col];
  }
  float tot = block_reduce_sum256(part, red);
  if (tid == 0) out[b] = tot + ob[0];
}

// ---------------- launch ----------------
extern "C" void kernel_launch(void* const* d_in, const int* in_sizes, int n_in,
                              void* d_out, int out_size, void* d_ws, size_t ws_size,
                              hipStream_t stream) {
  const int*   source_ids   = (const int*)d_in[0];
  const int*   position_idx = (const int*)d_in[1];
  const unsigned char* attn_mask = (const unsigned char*)d_in[2];
  const float* word_emb = (const float*)d_in[3];
  const float* pos_emb  = (const float*)d_in[4];
  const float* ln_g     = (const float*)d_in[5];
  const float* ln_b     = (const float*)d_in[6];
  const float* Wqkv     = (const float*)d_in[7];
  const float* bqkv     = (const float*)d_in[8];
  const float* Wo       = (const float*)d_in[9];
  const float* bo       = (const float*)d_in[10];
  const float* ln1_g    = (const float*)d_in[11];
  const float* ln1_b    = (const float*)d_in[12];
  const float* W1       = (const float*)d_in[13];
  const float* b1       = (const float*)d_in[14];
  const float* W2       = (const float*)d_in[15];
  const float* b2       = (const float*)d_in[16];
  const float* ln2_g    = (const float*)d_in[17];
  const float* ln2_b    = (const float*)d_in[18];
  const float* dense_W  = (const float*)d_in[19];
  const float* dense_b  = (const float*)d_in[20];
  const float* out_W    = (const float*)d_in[21];
  const float* out_b    = (const float*)d_in[22];
  float* out = (float*)d_out;

  char* ws = (char*)d_ws;
  size_t off = 0;
  auto take = [&](size_t bytes) -> char* {
    char* p = ws + off;
    off += (bytes + 255) & ~(size_t)255;
    return p;
  };
  float*    h32    = (float*)   take((size_t)M_ * H_ * 4);
  _Float16* h16    = (_Float16*)take((size_t)M_ * H_ * 2);
  _Float16* qkv16  = (_Float16*)take((size_t)M_ * 3 * H_ * 2);
  _Float16* ctx16  = (_Float16*)take((size_t)M_ * H_ * 2);
  float*    proj32 = (float*)   take((size_t)M_ * H_ * 4);
  _Float16* ffn16  = (_Float16*)take((size_t)M_ * FF_ * 2);
  _Float16* vt16   = (_Float16*)take((size_t)B_ * NH_ * DH_ * S_ * 2);
  _Float16* wqkvT  = (_Float16*)take((size_t)L_ * H_ * 3 * H_ * 2);   // [3H][H]
  _Float16* woT    = (_Float16*)take((size_t)L_ * H_ * H_ * 2);       // [H][H]
  _Float16* w1T    = (_Float16*)take((size_t)L_ * H_ * FF_ * 2);      // [FF][H]
  _Float16* w2T    = (_Float16*)take((size_t)L_ * FF_ * H_ * 2);      // [H][FF]

  // one-shot f16 conversion + transpose of all weights (deterministic per call)
  for (int l = 0; l < L_; ++l) {
    convert_transpose_kernel<<<dim3(3 * H_ / 32, H_ / 32), 256, 0, stream>>>(
        Wqkv + (size_t)l * H_ * 3 * H_, wqkvT + (size_t)l * H_ * 3 * H_, H_, 3 * H_);
    convert_transpose_kernel<<<dim3(H_ / 32, H_ / 32), 256, 0, stream>>>(
        Wo + (size_t)l * H_ * H_, woT + (size_t)l * H_ * H_, H_, H_);
    convert_transpose_kernel<<<dim3(FF_ / 32, H_ / 32), 256, 0, stream>>>(
        W1 + (size_t)l * H_ * FF_, w1T + (size_t)l * H_ * FF_, H_, FF_);
    convert_transpose_kernel<<<dim3(H_ / 32, FF_ / 32), 256, 0, stream>>>(
        W2 + (size_t)l * FF_ * H_, w2T + (size_t)l * FF_ * H_, FF_, H_);
  }

  embed_ln_kernel<<<M_, 256, 0, stream>>>(source_ids, position_idx, attn_mask,
                                          word_emb, pos_emb, ln_g, ln_b, h32, h16);

  for (int l = 0; l < L_; ++l) {
    // QKV projection: [M,768] x [768,2304] -> f16
    gemm_kernel<0><<<dim3(3 * H_ / 128, M_ / 64), 256, 0, stream>>>(
        h16, wqkvT + (size_t)l * H_ * 3 * H_, bqkv + (size_t)l * 3 * H_,
        nullptr, qkv16, M_, 3 * H_, H_);
    // V transpose for WMMA B-operand access
    vtranspose_kernel<<<B_ * NH_, 256, 0, stream>>>(qkv16, vt16);
    // attention
    attention_kernel<<<B_ * NH_ * (S_ / 16), 32, 0, stream>>>(qkv16, vt16, attn_mask, ctx16);
    // output projection -> f32 (pre-residual)
    gemm_kernel<1><<<dim3(H_ / 128, M_ / 64), 256, 0, stream>>>(
        ctx16, woT + (size_t)l * H_ * H_, bo + (size_t)l * H_,
        proj32, nullptr, M_, H_, H_);
    ln_residual_kernel<<<M_, 256, 0, stream>>>(h32, h16, proj32,
                                               ln1_g + (size_t)l * H_, ln1_b + (size_t)l * H_);
    // FFN up + GELU -> f16
    gemm_kernel<2><<<dim3(FF_ / 128, M_ / 64), 256, 0, stream>>>(
        h16, w1T + (size_t)l * H_ * FF_, b1 + (size_t)l * FF_,
        nullptr, ffn16, M_, FF_, H_);
    // FFN down -> f32 (pre-residual)
    gemm_kernel<1><<<dim3(H_ / 128, M_ / 64), 256, 0, stream>>>(
        ffn16, w2T + (size_t)l * FF_ * H_, b2 + (size_t)l * H_,
        proj32, nullptr, M_, H_, FF_);
    ln_residual_kernel<<<M_, 256, 0, stream>>>(h32, h16, proj32,
                                               ln2_g + (size_t)l * H_, ln2_b + (size_t)l * H_);
  }

  head_kernel<<<B_, 256, 0, stream>>>(h32, dense_W, dense_b, out_W, out_b, out);
}
